// PaiNNInteraction_60601988547144
// MI455X (gfx1250) — matrix operand order repaired and checked by reading, compiled, not run on hardware
//
#include <hip/hip_runtime.h>

#define HDIM 128
#define RDIM 20
#define H3   384

typedef __attribute__((ext_vector_type(16))) __bf16 v16bf;
typedef __attribute__((ext_vector_type(8)))  __bf16 v8bf;
typedef __attribute__((ext_vector_type(8)))  float  v8f;
typedef __attribute__((ext_vector_type(4)))  int    i32x4;

// bf16 weight staging layout in d_ws (element offsets)
#define WS_FW1 0                       // 32 x 128  (K padded 20 -> 32 with zeros)
#define WS_SW1 (32 * 128)              // 128 x 128
#define WS_FW2 (WS_SW1 + 128 * 128)    // 128 x 384
#define WS_SW2 (WS_FW2 + 128 * 384)    // 128 x 384
#define WS_TOTAL (WS_SW2 + 128 * 384)  // elements
#define W1_ELEMS WS_FW2                // fW1pad + sW1 = 20480 elems
#define W1_BYTES (W1_ELEMS * 2)        // 40960 bytes staged per block

#if defined(__AMDGCN__) && __has_builtin(__builtin_amdgcn_global_load_async_to_lds_b128)
#define HAVE_ASYNC_LDS 1
#else
#define HAVE_ASYNC_LDS 0
#endif

#define WMMA_BF16(a, b, c) \
  __builtin_amdgcn_wmma_f32_16x16x32_bf16(false, (a), false, (b), (short)0, (c), false, false)

__device__ __forceinline__ float silu_f(float x) {
#if __has_builtin(__builtin_amdgcn_rcpf)
  return x * __builtin_amdgcn_rcpf(1.0f + __expf(-x));   // v_rcp_f32, no div fixup
#else
  return x / (1.0f + __expf(-x));
#endif
}

// A fragment (16x32 bf16) from a row-major f32 row; lane row = lane&15, kb = (lane>=16)*8
__device__ __forceinline__ v16bf a_from_f32(const float* __restrict__ rowp, int k0, int kb) {
  v16bf a;
#pragma unroll
  for (int i = 0; i < 8; ++i) {
    a[i]     = (__bf16)rowp[k0 + kb + i];
    a[i + 8] = (__bf16)rowp[k0 + 16 + kb + i];
  }
  return a;
}

__device__ __forceinline__ v16bf a_from_f32_clamped(const float* __restrict__ rowp, int kb, int kmax) {
  v16bf a;
#pragma unroll
  for (int i = 0; i < 8; ++i) {
    const int k1 = kb + i, k2 = 16 + kb + i;
    a[i]     = (k1 < kmax) ? (__bf16)rowp[k1] : (__bf16)0.0f;
    a[i + 8] = (k2 < kmax) ? (__bf16)rowp[k2] : (__bf16)0.0f;
  }
  return a;
}

// B fragment (32x16 bf16) from row-major [K,N] bf16 matrix (global OR LDS);
// lane column = n0 + (lane&15), K = k0 + (lane>=16)*8 + i and k0 + 16 + (lane>=16)*8 + i
__device__ __forceinline__ v16bf b_from_w(const __bf16* __restrict__ W, int ldn, int k0, int n0, int krow) {
  const int n = n0 + (krow & 15), kb = (krow >> 4) * 8;
  v16bf b;
#pragma unroll
  for (int i = 0; i < 8; ++i) {
    b[i]     = W[(size_t)(k0 + kb + i) * ldn + n];
    b[i + 8] = W[(size_t)(k0 + 16 + kb + i) * ldn + n];
  }
  return b;
}

// A fragment from a row-major 16x128 bf16 LDS tile
__device__ __forceinline__ v16bf a_from_lds(const __bf16* __restrict__ row, int k0, int kb) {
  const v8bf lo = *(const v8bf*)(row + k0 + kb);
  const v8bf hi = *(const v8bf*)(row + k0 + 16 + kb);
  v16bf a;
#pragma unroll
  for (int i = 0; i < 8; ++i) { a[i] = lo[i]; a[i + 8] = hi[i]; }
  return a;
}

__global__ void painn_prep_weights(const float* __restrict__ fW1, const float* __restrict__ sW1,
                                   const float* __restrict__ fW2, const float* __restrict__ sW2,
                                   __bf16* __restrict__ w) {
  const int i = blockIdx.x * blockDim.x + threadIdx.x;
  if (i < 32 * 128) {
    const int k = i / 128, n = i % 128;
    w[WS_FW1 + i] = (k < RDIM) ? (__bf16)fW1[k * 128 + n] : (__bf16)0.0f;
  }
  if (i < 128 * 128) w[WS_SW1 + i] = (__bf16)sW1[i];
  if (i < 128 * 384) {
    w[WS_FW2 + i] = (__bf16)fW2[i];
    w[WS_SW2 + i] = (__bf16)sW2[i];
  }
}

__global__ void painn_init_out(const float* __restrict__ s, const float* __restrict__ v,
                               float* __restrict__ out, int n_s, int n_v) {
  const int i = blockIdx.x * blockDim.x + threadIdx.x;
  if (i < n_s) out[i] = s[i];
  else if (i < n_s + n_v) out[i] = v[i - n_s];
}

__global__ void __launch_bounds__(128)
painn_edge_kernel(const float* __restrict__ s, const float* __restrict__ v,
                  const int* __restrict__ ei, const float* __restrict__ rbf,
                  const float* __restrict__ unit,
                  const float* __restrict__ fb1, const float* __restrict__ fb2,
                  const float* __restrict__ sb1, const float* __restrict__ sb2,
                  const __bf16* __restrict__ wbf,
                  float* __restrict__ out, int E, int N) {
  __shared__ __bf16 wlds[W1_ELEMS];        // 40 KB: block-staged fW1(pad) + sW1
  __shared__ __bf16 act[4][16 * HDIM];     // 16 KB: per-wave activation staging (reused hf -> hs)

  const int tid  = threadIdx.x;
  const int lane = tid & 31;
  const int wv   = tid >> 5;
  const int l16  = lane & 15;
  const int hlf  = lane >> 4;         // 0: lanes 0-15, 1: lanes 16-31
  const int kb   = hlf * 8;           // A-fragment K sub-base
  const int krow = l16 + hlf * 16;    // B-fragment K row

  const int tile = blockIdx.x * 4 + wv;
  const int base = tile * 16;

  // ---- cooperative stage of GEMM1 weights into LDS (async DMA path) ----
#if HAVE_ASYNC_LDS
  {
    __attribute__((address_space(1))) i32x4* g =
        (__attribute__((address_space(1))) i32x4*)(__bf16*)wbf;
    __attribute__((address_space(3))) i32x4* l =
        (__attribute__((address_space(3))) i32x4*)&wlds[0];
#pragma unroll 1
    for (int i = tid; i < W1_BYTES / 16; i += 128)
      __builtin_amdgcn_global_load_async_to_lds_b128(g + i, l + i, 0, 0);
#if __has_builtin(__builtin_amdgcn_s_wait_asynccnt)
    __builtin_amdgcn_s_wait_asynccnt(0);
#else
    asm volatile("s_wait_asynccnt 0" ::: "memory");
#endif
  }
#else
#pragma unroll 1
  for (int i = tid * 8; i < W1_ELEMS; i += 128 * 8)
    *(v8bf*)&wlds[i] = *(const v8bf*)&wbf[i];
#endif
  __syncthreads();

  if (base >= E) return;              // wave-uniform exit (WMMA needs full EXEC)

  float* out_s = out;
  float* out_v = out + (size_t)N * HDIM;

  const __bf16* fW1l = wlds + WS_FW1;
  const __bf16* sW1l = wlds + WS_SW1;
  const __bf16* fW2b = wbf + WS_FW2;
  const __bf16* sW2b = wbf + WS_SW2;

  // warm GEMM2 weight cachelines (global_prefetch_b8)
  __builtin_prefetch((const char*)fW2b + ((size_t)tid * 1504) % ((size_t)2 * 128 * 384 * 2), 0, 1);

  // Per-lane edge metadata.
  // A-layout row for GEMM1: row m = l16.  D-layout rows: m = hlf*8 + r.
  const int eA   = (base + l16 < E) ? (base + l16) : (E - 1);
  const int srcA = ei[eA];

  int eD[8], srcD[8], dstD[8];
  float unitD[8][3];
#pragma unroll
  for (int r = 0; r < 8; ++r) {
    const int e  = base + hlf * 8 + r;
    const int ec = (e < E) ? e : (E - 1);
    eD[r]   = e;
    srcD[r] = ei[ec];
    dstD[r] = ei[E + ec];
#pragma unroll
    for (int d = 0; d < 3; ++d) unitD[r][d] = unit[(size_t)ec * 3 + d];
  }

  __bf16* abuf = &act[wv][0];
  const __bf16* arow = abuf + l16 * HDIM;

  v16bf hfA[4], hsA[4];

  // ---- GEMM1a: hf = silu(rbf @ fW1 + fb1) through the staging buffer ----
  {
    const float* rrow = rbf + (size_t)eA * RDIM;
    const v16bf aR = a_from_f32_clamped(rrow, kb, RDIM);   // K padded 20 -> 32
#pragma unroll 1
    for (int nt = 0; nt < 8; ++nt) {
      const int n0 = nt * 16;
      v8f acc = {};
      acc = WMMA_BF16(aR, b_from_w(fW1l, HDIM, 0, n0, krow), acc);
      const float b = fb1[n0 + l16];
#pragma unroll
      for (int r = 0; r < 8; ++r)
        abuf[(hlf * 8 + r) * HDIM + n0 + l16] = (__bf16)silu_f(acc[r] + b);
    }
#pragma unroll
    for (int kc = 0; kc < 4; ++kc) hfA[kc] = a_from_lds(arow, kc * 32, kb);
  }

  // ---- GEMM1b: hs = silu(s[src] @ sW1 + sb1), reusing the same buffer ----
  {
    const float* srow = s + (size_t)srcA * HDIM;
    v16bf aS[4];
#pragma unroll
    for (int kc = 0; kc < 4; ++kc) aS[kc] = a_from_f32(srow, kc * 32, kb);
#pragma unroll 1
    for (int nt = 0; nt < 8; ++nt) {
      const int n0 = nt * 16;
      v8f acc = {};
#pragma unroll
      for (int kc = 0; kc < 4; ++kc)
        acc = WMMA_BF16(aS[kc], b_from_w(sW1l, HDIM, kc * 32, n0, krow), acc);
      const float b = sb1[n0 + l16];
#pragma unroll
      for (int r = 0; r < 8; ++r)
        abuf[(hlf * 8 + r) * HDIM + n0 + l16] = (__bf16)silu_f(acc[r] + b);
    }
#pragma unroll
    for (int kc = 0; kc < 4; ++kc) hsA[kc] = a_from_lds(arow, kc * 32, kb);
  }

  // ---- GEMM2 part 1: ds columns (msg[:, 0:128]) -> scatter into out_s ----
#pragma unroll 1
  for (int nt = 0; nt < 8; ++nt) {
    const int n0 = nt * 16;
    v8f aF = {}, aSc = {};
#pragma unroll
    for (int kc = 0; kc < 4; ++kc) {
      aF  = WMMA_BF16(hfA[kc], b_from_w(fW2b, H3, kc * 32, n0, krow), aF);
      aSc = WMMA_BF16(hsA[kc], b_from_w(sW2b, H3, kc * 32, n0, krow), aSc);
    }
    const float bf2 = fb2[n0 + l16];
    const float bs2 = sb2[n0 + l16];
    const int col = n0 + l16;
#pragma unroll
    for (int r = 0; r < 8; ++r) {
      if (eD[r] < E) {
        const float val = (aF[r] + bf2) * (aSc[r] + bs2);
        atomicAdd(&out_s[(size_t)dstD[r] * HDIM + col], val);
      }
    }
  }

  // ---- GEMM2 part 2: dv_vec (cols 128..255) & dv_rad (cols 256..383) -> out_v ----
#pragma unroll 1
  for (int ct = 0; ct < 8; ++ct) {
    const int n0v = HDIM + ct * 16;
    const int n0r = 2 * HDIM + ct * 16;
    v8f aFv = {}, aSv = {}, aFr = {}, aSr = {};
#pragma unroll
    for (int kc = 0; kc < 4; ++kc) {
      aFv = WMMA_BF16(hfA[kc], b_from_w(fW2b, H3, kc * 32, n0v, krow), aFv);
      aSv = WMMA_BF16(hsA[kc], b_from_w(sW2b, H3, kc * 32, n0v, krow), aSv);
      aFr = WMMA_BF16(hfA[kc], b_from_w(fW2b, H3, kc * 32, n0r, krow), aFr);
      aSr = WMMA_BF16(hsA[kc], b_from_w(sW2b, H3, kc * 32, n0r, krow), aSr);
    }
    const float bfv = fb2[n0v + l16], bsv = sb2[n0v + l16];
    const float bfr = fb2[n0r + l16], bsr = sb2[n0r + l16];
    const int h = ct * 16 + l16;
#pragma unroll
    for (int r = 0; r < 8; ++r) {
      if (eD[r] >= E) continue;
      const float mv = (aFv[r] + bfv) * (aSv[r] + bsv);   // dv_vec coefficient
      const float mr = (aFr[r] + bfr) * (aSr[r] + bsr);   // dv_rad coefficient
      const size_t vsrc = (size_t)srcD[r] * 3 * HDIM;
      const size_t vdst = (size_t)dstD[r] * 3 * HDIM;
#pragma unroll
      for (int d = 0; d < 3; ++d) {
        const float contrib = mv * v[vsrc + d * HDIM + h] + mr * unitD[r][d];
        atomicAdd(&out_v[vdst + d * HDIM + h], contrib);
      }
    }
  }
}

extern "C" void kernel_launch(void* const* d_in, const int* in_sizes, int n_in,
                              void* d_out, int out_size, void* d_ws, size_t ws_size,
                              hipStream_t stream) {
  const float* s    = (const float*)d_in[0];
  const float* v    = (const float*)d_in[1];
  const int*   ei   = (const int*)d_in[2];
  const float* rbf  = (const float*)d_in[3];
  const float* unit = (const float*)d_in[4];
  const float* fW1  = (const float*)d_in[5];
  const float* fb1  = (const float*)d_in[6];
  const float* fW2  = (const float*)d_in[7];
  const float* fb2  = (const float*)d_in[8];
  const float* sW1  = (const float*)d_in[9];
  const float* sb1  = (const float*)d_in[10];
  const float* sW2  = (const float*)d_in[11];
  const float* sb2  = (const float*)d_in[12];
  float* out  = (float*)d_out;
  __bf16* wbf = (__bf16*)d_ws;

  const int N = in_sizes[0] / HDIM;   // 16384
  const int E = in_sizes[3] / RDIM;   // 400000

  // 1) stage bf16 weights in d_ws
  painn_prep_weights<<<(128 * 384 + 255) / 256, 256, 0, stream>>>(fW1, sW1, fW2, sW2, wbf);

  // 2) out = concat(s, v)
  const int tot = N * HDIM + N * 3 * HDIM;
  painn_init_out<<<(tot + 255) / 256, 256, 0, stream>>>(s, v, out, N * HDIM, N * 3 * HDIM);

  // 3) edge tiles: 16 edges per wave, 4 waves per block
  const int tiles  = (E + 15) / 16;
  const int blocks = (tiles + 3) / 4;
  painn_edge_kernel<<<blocks, 128, 0, stream>>>(s, v, ei, rbf, unit,
                                                fb1, fb2, sb1, sb2, wbf, out, E, N);
}
